// ForgetMemModule_33612414058554
// MI455X (gfx1250) — matrix operand order repaired and checked by reading, not compiled, not run on hardware
//
// CDNA5 / gfx1250 (MI455X) implementation of ForgetMemModule scan.
//
// Design:
//  * kernel 1 (x8): fp32 -> bf16 convert + transpose of all weight matrices into d_ws
//    (Wt[N][K] so WMMA B-fragments are K-contiguous 2x16B/lane loads).
//  * kernel 2: GX = hist @ Ww + Bw for all B*T rows (full-M WMMA GEMM), bf16 in d_ws.
//    This hoists the M=1 input-gate projection out of the sequential scan.
//  * kernel 3: persistent scan. 128 blocks (one per batch), 256 threads (8 wave32).
//    State mem[10][1024] fp32 lives in LDS for all 64 steps. Per step, every
//    [<=16 x 1024] x [1024 x N] GEMM runs on v_wmma_f32_16x16x32_bf16: 8 waves x
//    8 column tiles, k-loop reuses one A fragment (LDS) across 8 B tiles (global/L2).
//    Tiny attention (10x11 softmax) and gates run on VALU.
//
// Workspace: 6*2MB (HxH) + 2*4MB (Hx2H) + 32MB (GX) = 52MB, fully rewritten per call.
#include <hip/hip_runtime.h>

typedef unsigned short u16;
typedef unsigned int   u32;
typedef __attribute__((ext_vector_type(16))) __bf16 v16bf;
typedef __attribute__((ext_vector_type(8)))  float  v8f;
typedef __attribute__((ext_vector_type(4)))  u32    u32x4;

union FragU { v16bf v; u32x4 q[2]; };

__device__ __forceinline__ u16 f2bf(float f) {
  u32 u = __builtin_bit_cast(u32, f);
  u32 r = (u + 0x7FFFu + ((u >> 16) & 1u)) >> 16;   // RNE
  return (u16)r;
}
__device__ __forceinline__ float bf2f(u16 h) {
  u32 u = ((u32)h) << 16;
  return __builtin_bit_cast(float, u);
}

__device__ __forceinline__ void zero8(v8f* a) {
  v8f z = {0.f, 0.f, 0.f, 0.f, 0.f, 0.f, 0.f, 0.f};
#pragma unroll
  for (int j = 0; j < 8; ++j) a[j] = z;
}

// acc[j] += A(16x1024 bf16 in LDS) * Bt-columns for 8 consecutive 16-wide tiles.
// A fragment layout (16-bit A 16x32): lanes 0-15 rows 0-15 hold K[0..7],K[16..23];
// lanes 16-31 hold K[8..15],K[24..31].  B (32x16): lanes 0-15 col n K[0..7],K[16..23],
// lanes 16-31 col n K[8..15],K[24..31]; Bt row n is K-contiguous (pre-transposed weights).
__device__ __forceinline__ void gemm_row16_t8(const u16* __restrict__ Albs,
                                              const u16* __restrict__ Bt,
                                              int tile_base, int m, int half,
                                              v8f* acc) {
  const u16* arow = Albs + m * 1024;
#pragma unroll 1
  for (int kk = 0; kk < 32; ++kk) {
    FragU a;
    const int a0 = kk * 32 + half * 8;
    a.q[0] = *(const u32x4*)(arow + a0);
    a.q[1] = *(const u32x4*)(arow + a0 + 16);
    const int boff = kk * 32 + half * 8;
#pragma unroll
    for (int j = 0; j < 8; ++j) {
      FragU bb;
      const u16* bp = Bt + (size_t)((tile_base + j) * 16 + m) * 1024 + boff;
      bb.q[0] = *(const u32x4*)(bp);
      bb.q[1] = *(const u32x4*)(bp + 16);
      acc[j] = __builtin_amdgcn_wmma_f32_16x16x32_bf16(
          false, a.v, false, bb.v, (short)0, acc[j], false, false);
    }
  }
}

// --------------------------------------------------------------------------
// Kernel 1: fp32 [K][N] -> bf16 transposed [N][K]
__global__ void cvt_transpose_kernel(const float* __restrict__ W,
                                     u16* __restrict__ Wt, int N, int K) {
  size_t idx = (size_t)blockIdx.x * blockDim.x + threadIdx.x;
  if (idx < (size_t)K * N) {
    size_t k = idx / (size_t)N, n = idx % (size_t)N;
    Wt[n * (size_t)K + k] = f2bf(W[idx]);
  }
}

// --------------------------------------------------------------------------
// Kernel 2: GX[g][0..2047] = hist[g] @ Ww + Bw  (bf16 out), g = 0..8191
__global__ __launch_bounds__(256) void gx_precompute_kernel(
    const float* __restrict__ hist, const u16* __restrict__ Wwt,
    const float* __restrict__ Bw, u16* __restrict__ GX) {
  __shared__ u16 Abf[16 * 1024];
  const int tid = threadIdx.x, blk = blockIdx.x;
  const int wave = tid >> 5, lane = tid & 31, half = lane >> 4, m = lane & 15;
  const float* src = hist + (size_t)blk * 16 * 1024;
  for (int i = tid; i < 16 * 1024; i += 256) Abf[i] = f2bf(src[i]);
  __syncthreads();
#pragma unroll 1
  for (int pass = 0; pass < 2; ++pass) {
    v8f acc[8];
    zero8(acc);
    const int tb = wave * 16 + pass * 8;
    gemm_row16_t8(Abf, Wwt, tb, m, half, acc);
#pragma unroll
    for (int j = 0; j < 8; ++j)
#pragma unroll
      for (int r = 0; r < 8; ++r) {
        int row = r + 8 * half;
        int col = (tb + j) * 16 + m;
        GX[((size_t)blk * 16 + row) * 2048 + col] = f2bf(acc[j][r] + Bw[col]);
      }
  }
}

// --------------------------------------------------------------------------
// Kernel 3: persistent per-batch scan over T=64 steps.
__global__ __launch_bounds__(256) void forget_mem_scan_kernel(
    const float* __restrict__ hist, const float* __restrict__ mem0,
    const u16* __restrict__ wqt, const u16* __restrict__ wkt,
    const u16* __restrict__ wvt, const u16* __restrict__ wot,
    const u16* __restrict__ w1t, const u16* __restrict__ w2t,
    const u16* __restrict__ uwt, const u16* __restrict__ gx,
    const float* __restrict__ bq, const float* __restrict__ bk,
    const float* __restrict__ bv, const float* __restrict__ bo,
    const float* __restrict__ b1, const float* __restrict__ b2,
    const float* __restrict__ bu, float* __restrict__ out) {
  extern __shared__ char smem_raw[];
  float* memf  = (float*)smem_raw;            // [10][1024] fp32 master state
  float* nmf   = memf + 10240;                // [10][1024] fp32 nm accumulator
  u16*   membf = (u16*)(nmf + 10240);         // [16][1024] bf16: kv A / attn_out A
  u16*   Xb    = membf + 16384;               // [16][1024] bf16: q -> v -> nm -> tanh(mem)
  u16*   Yb    = Xb + 16384;                  // [16][1024] bf16: k -> h1
  float* sc    = (float*)(Yb + 16384);        // [10][12] scores / attn weights

  const int tid = threadIdx.x;
  const int b   = blockIdx.x;
  const int wave = tid >> 5, lane = tid & 31, half = lane >> 4, m = lane & 15;
  const int tb0 = wave * 8;

  for (int i = tid; i < 10240; i += 256) memf[i] = mem0[(size_t)b * 10240 + i];
  for (int i = tid; i < 16384; i += 256) membf[i] = 0;   // zeros pad rows 11..15
  __syncthreads();

#pragma unroll 1
  for (int t = 0; t < 64; ++t) {
    const float* xrow = hist + ((size_t)b * 64 + t) * 1024;
    // kv = [mem ; x_t] in bf16 (rows 0..9 mem, row 10 x_t, rows 11..15 stay 0)
    for (int i = tid; i < 11264; i += 256) {
      int row = i >> 10, col = i & 1023;
      membf[i] = (row < 10) ? f2bf(memf[i]) : f2bf(xrow[col]);
    }
    __syncthreads();

    // ---- q = (mem @ wq + bq) * H^-0.5  -> Xb
    {
      v8f acc[8]; zero8(acc);
      gemm_row16_t8(membf, wqt, tb0, m, half, acc);
#pragma unroll
      for (int j = 0; j < 8; ++j)
#pragma unroll
        for (int r = 0; r < 8; ++r) {
          int row = r + 8 * half, col = (tb0 + j) * 16 + m;
          Xb[row * 1024 + col] = f2bf((acc[j][r] + bq[col]) * 0.03125f);
        }
    }
    // ---- k = kv @ wk + bk  -> Yb
    {
      v8f acc[8]; zero8(acc);
      gemm_row16_t8(membf, wkt, tb0, m, half, acc);
#pragma unroll
      for (int j = 0; j < 8; ++j)
#pragma unroll
        for (int r = 0; r < 8; ++r) {
          int row = r + 8 * half, col = (tb0 + j) * 16 + m;
          Yb[row * 1024 + col] = f2bf(acc[j][r] + bk[col]);
        }
    }
    __syncthreads();

    // ---- scores[s][tt] = q[s] . k[tt]  (110 dot products, one per wave slot)
    for (int p = wave; p < 110; p += 8) {
      int s = p / 11, tt = p % 11;
      const u16* qp = Xb + s * 1024;
      const u16* kp = Yb + tt * 1024;
      float a = 0.f;
#pragma unroll 4
      for (int i = 0; i < 32; ++i) {
        int h = i * 32 + lane;
        a += bf2f(qp[h]) * bf2f(kp[h]);
      }
#pragma unroll
      for (int off = 16; off > 0; off >>= 1) a += __shfl_xor(a, off, 32);
      if (lane == 0) sc[s * 12 + tt] = a;
    }
    __syncthreads();
    // ---- softmax over 11 kv slots (one thread per memory slot)
    if (tid < 10) {
      float mx = -1e30f;
      for (int tt = 0; tt < 11; ++tt) mx = fmaxf(mx, sc[tid * 12 + tt]);
      float ssum = 0.f, e[11];
      for (int tt = 0; tt < 11; ++tt) { e[tt] = expf(sc[tid * 12 + tt] - mx); ssum += e[tt]; }
      float inv = 1.f / ssum;
      for (int tt = 0; tt < 11; ++tt) sc[tid * 12 + tt] = e[tt] * inv;
    }
    __syncthreads();

    // ---- v = kv @ wv + bv  -> Xb (q is dead)
    {
      v8f acc[8]; zero8(acc);
      gemm_row16_t8(membf, wvt, tb0, m, half, acc);
#pragma unroll
      for (int j = 0; j < 8; ++j)
#pragma unroll
        for (int r = 0; r < 8; ++r) {
          int row = r + 8 * half, col = (tb0 + j) * 16 + m;
          Xb[row * 1024 + col] = f2bf(acc[j][r] + bv[col]);
        }
    }
    __syncthreads();

    // ---- attn_out_raw = attn @ v  -> membf rows 0..9 (kv A is dead)
    for (int i = tid; i < 10240; i += 256) {
      int s = i >> 10, h = i & 1023;
      float a = 0.f;
#pragma unroll
      for (int tt = 0; tt < 11; ++tt) a += sc[s * 12 + tt] * bf2f(Xb[tt * 1024 + h]);
      membf[i] = f2bf(a);
    }
    __syncthreads();

    // ---- nm = mem + attn_out_raw @ wo + bo  -> nmf (fp32) and Xb (bf16 A for w1)
    {
      v8f acc[8]; zero8(acc);
      gemm_row16_t8(membf, wot, tb0, m, half, acc);
#pragma unroll
      for (int j = 0; j < 8; ++j)
#pragma unroll
        for (int r = 0; r < 8; ++r) {
          int row = r + 8 * half, col = (tb0 + j) * 16 + m;
          if (row < 10) {
            float nm = memf[row * 1024 + col] + acc[j][r] + bo[col];
            nmf[row * 1024 + col] = nm;
            Xb[row * 1024 + col] = f2bf(nm);
          } else {
            Xb[row * 1024 + col] = 0;   // clean pad rows for the MLP A operand
          }
        }
    }
    __syncthreads();

    // ---- h1 = relu(nm @ w1 + b1)  -> Yb
    {
      v8f acc[8]; zero8(acc);
      gemm_row16_t8(Xb, w1t, tb0, m, half, acc);
#pragma unroll
      for (int j = 0; j < 8; ++j)
#pragma unroll
        for (int r = 0; r < 8; ++r) {
          int row = r + 8 * half, col = (tb0 + j) * 16 + m;
          Yb[row * 1024 + col] = f2bf(fmaxf(acc[j][r] + b1[col], 0.f));
        }
    }
    __syncthreads();

    // ---- nm += relu(h1 @ w2 + b2)
    {
      v8f acc[8]; zero8(acc);
      gemm_row16_t8(Yb, w2t, tb0, m, half, acc);
#pragma unroll
      for (int j = 0; j < 8; ++j)
#pragma unroll
        for (int r = 0; r < 8; ++r) {
          int row = r + 8 * half, col = (tb0 + j) * 16 + m;
          if (row < 10) nmf[row * 1024 + col] += fmaxf(acc[j][r] + b2[col], 0.f);
        }
    }
    __syncthreads();

    // ---- tanh(mem) -> Xb rows 0..9 (rows 10..15 already 0 from nm epilogue)
    for (int i = tid; i < 10240; i += 256) Xb[i] = f2bf(tanhf(memf[i]));
    __syncthreads();

    // ---- gates: [i|f] = sigmoid(tanh(mem) @ Uw + Bu + GX[b,t]); update mem in place.
    const u16* gxr = gx + ((size_t)b * 64 + t) * 2048;
#pragma unroll 1
    for (int pass = 0; pass < 2; ++pass) {
      const int tbg = wave * 8 + pass * 4;
      v8f ai[4], af[4];
      {
        v8f z = {0.f, 0.f, 0.f, 0.f, 0.f, 0.f, 0.f, 0.f};
#pragma unroll
        for (int p = 0; p < 4; ++p) { ai[p] = z; af[p] = z; }
      }
      const u16* arow = Xb + m * 1024;
#pragma unroll 1
      for (int kk = 0; kk < 32; ++kk) {
        FragU a;
        const int a0 = kk * 32 + half * 8;
        a.q[0] = *(const u32x4*)(arow + a0);
        a.q[1] = *(const u32x4*)(arow + a0 + 16);
        const int boff = kk * 32 + half * 8;
#pragma unroll
        for (int p = 0; p < 4; ++p) {
          const int ni = (tbg + p) * 16 + m;          // input-gate column
          FragU bi, bf_;
          const u16* bpi = uwt + (size_t)ni * 1024 + boff;
          bi.q[0] = *(const u32x4*)bpi;
          bi.q[1] = *(const u32x4*)(bpi + 16);
          ai[p] = __builtin_amdgcn_wmma_f32_16x16x32_bf16(
              false, a.v, false, bi.v, (short)0, ai[p], false, false);
          const u16* bpf = uwt + (size_t)(ni + 1024) * 1024 + boff;  // forget col
          bf_.q[0] = *(const u32x4*)bpf;
          bf_.q[1] = *(const u32x4*)(bpf + 16);
          af[p] = __builtin_amdgcn_wmma_f32_16x16x32_bf16(
              false, a.v, false, bf_.v, (short)0, af[p], false, false);
        }
      }
#pragma unroll
      for (int p = 0; p < 4; ++p)
#pragma unroll
        for (int r = 0; r < 8; ++r) {
          int row = r + 8 * half;
          if (row < 10) {
            int col = (tbg + p) * 16 + m;
            float gi = ai[p][r] + bu[col] + bf2f(gxr[col]);
            float gf = af[p][r] + bu[col + 1024] + bf2f(gxr[col + 1024]);
            gi = 1.f / (1.f + expf(-gi));
            gf = 1.f / (1.f + expf(-gf));
            int idx = row * 1024 + col;
            memf[idx] = gi * tanhf(nmf[idx]) + gf * memf[idx];
          }
        }
    }
    __syncthreads();
  }

  for (int i = tid; i < 10240; i += 256) out[(size_t)b * 10240 + i] = memf[i];
}

// --------------------------------------------------------------------------
extern "C" void kernel_launch(void* const* d_in, const int* in_sizes, int n_in,
                              void* d_out, int out_size, void* d_ws, size_t ws_size,
                              hipStream_t stream) {
  const float* hist = (const float*)d_in[0];   // [8192,1024]
  const float* mem0 = (const float*)d_in[1];   // [128,10,1024]
  const float* wq = (const float*)d_in[2];  const float* bq = (const float*)d_in[3];
  const float* wk = (const float*)d_in[4];  const float* bk = (const float*)d_in[5];
  const float* wv = (const float*)d_in[6];  const float* bv = (const float*)d_in[7];
  const float* wo = (const float*)d_in[8];  const float* bo = (const float*)d_in[9];
  const float* w1 = (const float*)d_in[10]; const float* b1 = (const float*)d_in[11];
  const float* w2 = (const float*)d_in[12]; const float* b2 = (const float*)d_in[13];
  const float* Ww = (const float*)d_in[14]; const float* Bw = (const float*)d_in[15];
  const float* Uw = (const float*)d_in[16]; const float* Bu = (const float*)d_in[17];
  // d_in[18] = hist_conv_len: all equal to T, mask path unused.

  const size_t HH = (size_t)1024 * 1024;
  u16* ws  = (u16*)d_ws;
  u16* WQT = ws + 0 * HH;
  u16* WKT = ws + 1 * HH;
  u16* WVT = ws + 2 * HH;
  u16* WOT = ws + 3 * HH;
  u16* W1T = ws + 4 * HH;
  u16* W2T = ws + 5 * HH;
  u16* WWT = ws + 6 * HH;    // 2*HH elements ([2048][1024])
  u16* UWT = ws + 8 * HH;    // 2*HH elements
  u16* GXW = ws + 10 * HH;   // 16*HH elements ([8192][2048])  -> 52MB total

  const int blk1 = (int)((HH + 255) / 256);
  cvt_transpose_kernel<<<blk1, 256, 0, stream>>>(wq, WQT, 1024, 1024);
  cvt_transpose_kernel<<<blk1, 256, 0, stream>>>(wk, WKT, 1024, 1024);
  cvt_transpose_kernel<<<blk1, 256, 0, stream>>>(wv, WVT, 1024, 1024);
  cvt_transpose_kernel<<<blk1, 256, 0, stream>>>(wo, WOT, 1024, 1024);
  cvt_transpose_kernel<<<blk1, 256, 0, stream>>>(w1, W1T, 1024, 1024);
  cvt_transpose_kernel<<<blk1, 256, 0, stream>>>(w2, W2T, 1024, 1024);
  cvt_transpose_kernel<<<2 * blk1, 256, 0, stream>>>(Ww, WWT, 2048, 1024);
  cvt_transpose_kernel<<<2 * blk1, 256, 0, stream>>>(Uw, UWT, 2048, 1024);

  gx_precompute_kernel<<<512, 256, 0, stream>>>(hist, WWT, Bw, GXW);

  // LDS: 2*10240*4 (memf,nmf) + 3*16384*2 (membf,Xb,Yb) + 120*4 (scores)
  const size_t lds_bytes = 2 * 10240 * 4 + 3 * 16384 * 2 + 120 * 4;  // 180,704 B
  forget_mem_scan_kernel<<<128, 256, lds_bytes, stream>>>(
      hist, mem0, WQT, WKT, WVT, WOT, W1T, W2T, UWT, GXW,
      bq, bk, bv, bo, b1, b2, Bu, (float*)d_out);
}
